// TransformerEncoder_32985348833399
// MI455X (gfx1250) — matrix operand order repaired and checked by reading, not compiled, not run on hardware
//
#include <hip/hip_runtime.h>
#include <math.h>

// ---------------- problem constants (match reference) ----------------
#define BB   2
#define NN   2048
#define FF   1024
#define HIDC 1024
#define HEADC 16
#define DH   64        // HID/HEAD
#define MLPC 4096
#define ROWS (BB*NN)   // 4096

typedef __attribute__((ext_vector_type(16))) __bf16 bf16x16;
typedef __attribute__((ext_vector_type(8)))  float  floatx8;

// A-fragment K index for 16-bit WMMA (16x32 MxK): lanes 0-15 hold K {0..7,16..23},
// lanes 16-31 hold K {8..15,24..31}  (ISA 7.12.2)
__device__ __forceinline__ int a_kidx(int e, int hi) {
    return (e < 8) ? (hi * 8 + e) : (16 + hi * 8 + (e - 8));
}

__device__ __forceinline__ float gelu_exact(float v) {
    return 0.5f * v * (1.0f + erff(v * 0.70710678118654752f));
}

// ---------------- f32 -> bf16 conversion ----------------
__global__ void f2bf_kernel(const float* __restrict__ in, __bf16* __restrict__ out, int n) {
    int i = blockIdx.x * 256 + threadIdx.x;
    if (i < n) out[i] = (__bf16)in[i];
}

// ---------------- LayerNorm (row of FF=1024, 256 threads/row) -> bf16 ----------------
__global__ void ln_kernel(const float* __restrict__ x, const float* __restrict__ g,
                          const float* __restrict__ be, __bf16* __restrict__ out) {
    __shared__ float red[256];
    const int row = blockIdx.x;
    const int tid = threadIdx.x;
    const float* xr = x + (size_t)row * FF;
    float lv[4];
#pragma unroll
    for (int e = 0; e < 4; ++e) lv[e] = xr[tid * 4 + e];
    float s = lv[0] + lv[1] + lv[2] + lv[3];
    red[tid] = s;
    __syncthreads();
    for (int st = 128; st > 0; st >>= 1) {
        if (tid < st) red[tid] += red[tid + st];
        __syncthreads();
    }
    const float mu = red[0] * (1.0f / FF);
    __syncthreads();
    float vs = 0.f;
#pragma unroll
    for (int e = 0; e < 4; ++e) { float d = lv[e] - mu; vs += d * d; }
    red[tid] = vs;
    __syncthreads();
    for (int st = 128; st > 0; st >>= 1) {
        if (tid < st) red[tid] += red[tid + st];
        __syncthreads();
    }
    const float rstd = rsqrtf(red[0] * (1.0f / FF) + 1e-5f);
#pragma unroll
    for (int e = 0; e < 4; ++e) {
        int idx = tid * 4 + e;
        out[(size_t)row * FF + idx] = (__bf16)((lv[e] - mu) * rstd * g[idx] + be[idx]);
    }
}

// ---------------- generic WMMA GEMM: C[M,N] = act(A[M,K] * W[N,K]^T + bias) (+res) ------
// block = 256 threads = 8 waves laid out 4(M) x 2(N). Each wave owns a 32x64 macro-tile
// (2x4 WMMA tiles, 8 f32 accumulators) -> 8 v_wmma per wave per K-step, 12 LDS b128
// fragment loads. Workgroup tile 128(M) x 128(N); K step 32; LDS 20 KB.
// Epilogue configuration is compile-time to keep it branch-free.
template <int ACT, bool HAS_BIAS, bool HAS_RES, bool OUT_F, bool OUT_B>
__global__ void gemm_bf16_kernel(const __bf16* __restrict__ A,   // [M,K] row-major
                                 const __bf16* __restrict__ W,   // [N,K] row-major
                                 const float*  __restrict__ bias,// [N] or null
                                 const float*  __restrict__ res, // [M,N] or null
                                 float*        __restrict__ outF,// [M,N] or null
                                 __bf16*       __restrict__ outB,// [M,N] or null
                                 int M, int N, int K) {
    __shared__ __bf16 Asl[128][40];  // 80B row stride (16B aligned)
    __shared__ __bf16 Wsl[128][40];

    const int tid  = threadIdx.x;
    const int wave = tid >> 5;
    const int lane = tid & 31;
    const int wm   = wave >> 1;        // 0..3  (M macro-row, 32 rows each)
    const int wn   = wave & 1;         // 0..1  (N macro-col, 64 cols each)
    const int c    = lane & 15;
    const int hi   = lane >> 4;
    const int tileM = blockIdx.x * 128;
    const int tileN = blockIdx.y * 128;

    // staging: A 128x32 and W 128x32, 16 elems (two b128) per thread each
    const int srow = tid >> 1;             // 0..127
    const int scol = (tid & 1) * 16;       // 0,16

    floatx8 acc[2][4] = {};

    for (int k0 = 0; k0 < K; k0 += 32) {
        const __bf16* ap = A + (size_t)(tileM + srow) * K + k0 + scol;
        uint4 av0 = *(const uint4*)(ap);
        uint4 av1 = *(const uint4*)(ap + 8);
        *(uint4*)&Asl[srow][scol]     = av0;
        *(uint4*)&Asl[srow][scol + 8] = av1;
        const __bf16* wp = W + (size_t)(tileN + srow) * K + k0 + scol;
        uint4 wv0 = *(const uint4*)(wp);
        uint4 wv1 = *(const uint4*)(wp + 8);
        *(uint4*)&Wsl[srow][scol]     = wv0;
        *(uint4*)&Wsl[srow][scol + 8] = wv1;
        if (k0 + 32 < K) {
            __builtin_prefetch(ap + 32, 0, 3);
            __builtin_prefetch(wp + 32, 0, 3);
        }
        __syncthreads();

        bf16x16 af[2], bfr[4];
#pragma unroll
        for (int tm = 0; tm < 2; ++tm)
#pragma unroll
            for (int e = 0; e < 16; ++e)
                af[tm][e] = Asl[wm * 32 + tm * 16 + c][hi * 16 + e];
#pragma unroll
        for (int tn = 0; tn < 4; ++tn)
#pragma unroll
            for (int e = 0; e < 16; ++e)
                bfr[tn][e] = Wsl[wn * 64 + tn * 16 + c][hi * 16 + e];

#pragma unroll
        for (int tm = 0; tm < 2; ++tm)
#pragma unroll
            for (int tn = 0; tn < 4; ++tn)
                acc[tm][tn] = __builtin_amdgcn_wmma_f32_16x16x32_bf16(
                    false, af[tm], false, bfr[tn], (short)0, acc[tm][tn], false, false);
        __syncthreads();
    }

#pragma unroll
    for (int tn = 0; tn < 4; ++tn) {
        const int col = tileN + wn * 64 + tn * 16 + c;
        const float bval = HAS_BIAS ? bias[col] : 0.0f;
#pragma unroll
        for (int tm = 0; tm < 2; ++tm) {
#pragma unroll
            for (int i = 0; i < 8; ++i) {
                const int row = tileM + wm * 32 + tm * 16 + hi * 8 + i;
                float v = acc[tm][tn][i] + bval;
                if (ACT == 1) v = gelu_exact(v);
                const size_t idx = (size_t)row * N + col;
                if (HAS_RES) v += res[idx];
                if (OUT_F)   outF[idx] = v;
                if (OUT_B)   outB[idx] = (__bf16)v;
            }
        }
    }
}

// ---------------- flash attention: per-wave 16 query rows, stream K/V in 32-row blocks ---
// grid: (N/128, B*HEAD), block 256 (8 waves). q,k,v,out: bf16 [B*N, HID], head at col h*64.
__global__ void attn_kernel(const __bf16* __restrict__ q, const __bf16* __restrict__ k,
                            const __bf16* __restrict__ v, __bf16* __restrict__ o) {
    __shared__ __bf16 Kl[32][72];       // K block [32 rows][64 d + pad]
    __shared__ __bf16 Vt[64][40];       // V block transposed [d][32 rows + pad]
    __shared__ __bf16 Pl[8][16][40];    // per-wave P tile [16 rows][32 cols + pad]

    const int bh   = blockIdx.y;
    const int b    = bh >> 4;
    const int h    = bh & 15;
    const int tid  = threadIdx.x;
    const int wave = tid >> 5;
    const int lane = tid & 31;
    const int c    = lane & 15;
    const int hi   = lane >> 4;
    const int qbase = blockIdx.x * 128 + wave * 16;

    // load Q fragments (dh=64 -> two K=32 steps), A-layout
    bf16x16 qa[2];
    {
        const size_t qoff = ((size_t)(b * NN + qbase + c)) * HIDC + h * DH;
#pragma unroll
        for (int t = 0; t < 2; ++t)
#pragma unroll
            for (int e = 0; e < 16; ++e)
                qa[t][e] = q[qoff + t * 32 + hi * 16 + e];
    }

    float mi[8], li[8];
    floatx8 acc[4] = {};
#pragma unroll
    for (int i = 0; i < 8; ++i) { mi[i] = -1e30f; li[i] = 0.0f; }

    const int krow = tid >> 3;            // 0..31
    const int kcol = (tid & 7) * 8;       // 0..56

    for (int j = 0; j < NN; j += 32) {
        // stage K (row-major) and V (transposed) blocks
        {
            const size_t goff = ((size_t)(b * NN + j + krow)) * HIDC + h * DH + kcol;
            uint4 kv = *(const uint4*)(k + goff);
            *(uint4*)&Kl[krow][kcol] = kv;
            uint4 vv = *(const uint4*)(v + goff);
            const __bf16* vp = (const __bf16*)&vv;
#pragma unroll
            for (int e = 0; e < 8; ++e) Vt[kcol + e][krow] = vp[e];
        }
        __syncthreads();

        // S = Q*K^T  (two 16x16 column tiles), K-dim = dh
        floatx8 s0 = {}, s1 = {};
#pragma unroll
        for (int t = 0; t < 2; ++t) {
            bf16x16 kb0, kb1;
#pragma unroll
            for (int e = 0; e < 16; ++e) kb0[e] = Kl[c][t * 32 + hi * 16 + e];
#pragma unroll
            for (int e = 0; e < 16; ++e) kb1[e] = Kl[16 + c][t * 32 + hi * 16 + e];
            s0 = __builtin_amdgcn_wmma_f32_16x16x32_bf16(false, qa[t], false, kb0,
                                                         (short)0, s0, false, false);
            s1 = __builtin_amdgcn_wmma_f32_16x16x32_bf16(false, qa[t], false, kb1,
                                                         (short)0, s1, false, false);
        }

        // online softmax over the 32 new columns (rows = 8*hi + i, cols across 16 lanes)
#pragma unroll
        for (int i = 0; i < 8; ++i) {
            float sv0 = s0[i] * 0.03125f;   // 1/sqrt(HID) = 1/32
            float sv1 = s1[i] * 0.03125f;
            float mx = fmaxf(sv0, sv1);
#pragma unroll
            for (int off = 1; off < 16; off <<= 1)
                mx = fmaxf(mx, __shfl_xor(mx, off));
            const float mnew = fmaxf(mi[i], mx);
            const float scl  = __expf(mi[i] - mnew);
            const float p0   = __expf(sv0 - mnew);
            const float p1   = __expf(sv1 - mnew);
            float prt = p0 + p1;
#pragma unroll
            for (int off = 1; off < 16; off <<= 1)
                prt += __shfl_xor(prt, off);
            li[i] = li[i] * scl + prt;
            mi[i] = mnew;
#pragma unroll
            for (int t = 0; t < 4; ++t) acc[t][i] *= scl;
            Pl[wave][hi * 8 + i][c]      = (__bf16)p0;
            Pl[wave][hi * 8 + i][16 + c] = (__bf16)p1;
        }

        // O += P * V  (K-dim = 32 block rows, N = 64 head dims -> 4 tiles)
        bf16x16 pa;
#pragma unroll
        for (int e = 0; e < 16; ++e) pa[e] = Pl[wave][c][hi * 16 + e];
#pragma unroll
        for (int t = 0; t < 4; ++t) {
            bf16x16 vb;
#pragma unroll
            for (int e = 0; e < 16; ++e) vb[e] = Vt[t * 16 + c][hi * 16 + e];
            acc[t] = __builtin_amdgcn_wmma_f32_16x16x32_bf16(false, pa, false, vb,
                                                             (short)0, acc[t], false, false);
        }
        __syncthreads();
    }

    // normalize and write: out[b, n, h*64 + d]
#pragma unroll
    for (int i = 0; i < 8; ++i) {
        const float inv = 1.0f / li[i];
        const size_t base = ((size_t)(b * NN + qbase + hi * 8 + i)) * HIDC + h * DH;
#pragma unroll
        for (int t = 0; t < 4; ++t)
            o[base + t * 16 + c] = (__bf16)(acc[t][i] * inv);
    }
}

// ------------------------------- host side -------------------------------
extern "C" void kernel_launch(void* const* d_in, const int* in_sizes, int n_in,
                              void* d_out, int out_size, void* d_ws, size_t ws_size,
                              hipStream_t stream) {
    (void)in_sizes; (void)n_in; (void)out_size; (void)ws_size;

    const float* x     = (const float*)d_in[0];
    const float* la1_g = (const float*)d_in[1];
    const float* la1_b = (const float*)d_in[2];
    const float* Wq    = (const float*)d_in[3];
    const float* Wk    = (const float*)d_in[4];
    const float* Wv    = (const float*)d_in[5];
    const float* Wo    = (const float*)d_in[6];
    const float* bo    = (const float*)d_in[7];
    const float* la2_g = (const float*)d_in[8];
    const float* la2_b = (const float*)d_in[9];
    const float* W1    = (const float*)d_in[10];
    const float* b1    = (const float*)d_in[11];
    const float* W2    = (const float*)d_in[12];
    const float* b2    = (const float*)d_in[13];
    float* y = (float*)d_out;

    // workspace layout (bytes)
    char* ws = (char*)d_ws;
    const size_t MB = 1024ull * 1024ull;
    __bf16* h_bf    = (__bf16*)(ws + 0 * MB);   // 8 MB  (reused for h2 after QKV)
    __bf16* q_bf    = (__bf16*)(ws + 8 * MB);   // 8 MB
    __bf16* k_bf    = (__bf16*)(ws + 16 * MB);  // 8 MB
    __bf16* v_bf    = (__bf16*)(ws + 24 * MB);  // 8 MB
    __bf16* attn_bf = (__bf16*)(ws + 32 * MB);  // 8 MB
    float*  out_f   = (float*) (ws + 40 * MB);  // 16 MB
    __bf16* m1_bf   = (__bf16*)(ws + 56 * MB);  // 32 MB
    __bf16* Wq_bf   = (__bf16*)(ws + 88 * MB);  // 2 MB
    __bf16* Wk_bf   = (__bf16*)(ws + 90 * MB);  // 2 MB
    __bf16* Wv_bf   = (__bf16*)(ws + 92 * MB);  // 2 MB
    __bf16* Wo_bf   = (__bf16*)(ws + 94 * MB);  // 2 MB
    __bf16* W1_bf   = (__bf16*)(ws + 96 * MB);  // 8 MB
    __bf16* W2_bf   = (__bf16*)(ws + 104 * MB); // 8 MB

    const int nW  = HIDC * FF;     // 1M
    const int nW1 = MLPC * FF;     // 4M
    f2bf_kernel<<<(nW  + 255) / 256, 256, 0, stream>>>(Wq, Wq_bf, nW);
    f2bf_kernel<<<(nW  + 255) / 256, 256, 0, stream>>>(Wk, Wk_bf, nW);
    f2bf_kernel<<<(nW  + 255) / 256, 256, 0, stream>>>(Wv, Wv_bf, nW);
    f2bf_kernel<<<(nW  + 255) / 256, 256, 0, stream>>>(Wo, Wo_bf, nW);
    f2bf_kernel<<<(nW1 + 255) / 256, 256, 0, stream>>>(W1, W1_bf, nW1);
    f2bf_kernel<<<(nW1 + 255) / 256, 256, 0, stream>>>(W2, W2_bf, nW1);

    // LN1
    ln_kernel<<<ROWS, 256, 0, stream>>>(x, la1_g, la1_b, h_bf);

    // QKV projections  (workgroup tile 128x128)
    dim3 gP(ROWS / 128, HIDC / 128);
    gemm_bf16_kernel<0, false, false, false, true><<<gP, 256, 0, stream>>>(
        h_bf, Wq_bf, nullptr, nullptr, nullptr, q_bf, ROWS, HIDC, FF);
    gemm_bf16_kernel<0, false, false, false, true><<<gP, 256, 0, stream>>>(
        h_bf, Wk_bf, nullptr, nullptr, nullptr, k_bf, ROWS, HIDC, FF);
    gemm_bf16_kernel<0, false, false, false, true><<<gP, 256, 0, stream>>>(
        h_bf, Wv_bf, nullptr, nullptr, nullptr, v_bf, ROWS, HIDC, FF);

    // flash attention
    attn_kernel<<<dim3(NN / 128, BB * HEADC), 256, 0, stream>>>(q_bf, k_bf, v_bf, attn_bf);

    // out-proj + bias + residual(x) -> out_f (fp32)
    gemm_bf16_kernel<0, true, true, true, false><<<gP, 256, 0, stream>>>(
        attn_bf, Wo_bf, bo, x, out_f, nullptr, ROWS, FF, HIDC);

    // LN2 (reuse h_bf)
    ln_kernel<<<ROWS, 256, 0, stream>>>(out_f, la2_g, la2_b, h_bf);

    // MLP1: gelu(h2 @ W1^T + b1) -> bf16
    dim3 gM1(ROWS / 128, MLPC / 128);
    gemm_bf16_kernel<1, true, false, false, true><<<gM1, 256, 0, stream>>>(
        h_bf, W1_bf, b1, nullptr, nullptr, m1_bf, ROWS, MLPC, FF);

    // MLP2: gelu(m1 @ W2^T + b2) + out -> final fp32 output
    gemm_bf16_kernel<1, true, true, true, false><<<gP, 256, 0, stream>>>(
        m1_bf, W2_bf, b2, out_f, y, nullptr, ROWS, FF, MLPC);
}
